// MaskBinActiveConv2d_25426206392807
// MI455X (gfx1250) — compile-verified, matrix operand order checked by reading, not run on hardware
//
#include <hip/hip_runtime.h>
#include <stdint.h>

typedef __attribute__((ext_vector_type(16))) _Float16 v16h;
typedef __attribute__((ext_vector_type(8)))  float    v8f;
typedef __attribute__((ext_vector_type(4)))  float    f32x4;
typedef __attribute__((ext_vector_type(4)))  uint32_t u32x4;
typedef __attribute__((ext_vector_type(8)))  uint32_t u32x8;

#define BATCH 32
#define CIN   256
#define COUT  256
#define WID   56
#define HW    3136           // 56*56
#define NPIX  (BATCH * HW)   // 100352
#define KDIM  (CIN * 9)      // 2304, ordered (kh,kw,ci): k = (kh*3+kw)*256 + ci
#define BM    128
#define BN    128
#define BK    32
#define NKT   (KDIM / BK)    // 72
#define ROWH  40             // BK + 8 pad halves -> 80B row; TDM pads 16DW+4DW

union FragU { f32x4 q[2]; v16h v; };

// ---------------------------------------------------------------------------
// Prepass: gate by sign(mask), convert to f16, reorder to [co][r][ci], r=kh*3+kw
// ---------------------------------------------------------------------------
__global__ __launch_bounds__(256)
void prep_weights(const float* __restrict__ W, const float* __restrict__ Mask,
                  _Float16* __restrict__ Wk)
{
    const int co = blockIdx.x;
    const int ci = threadIdx.x;
    const float m   = Mask[co];
    const float sgn = (m > 0.f) ? 1.f : ((m < 0.f) ? -1.f : 0.f);
    const float g   = 1.f + (sgn - 1.f) * 0.5f;     // 1 if mask>0, 0 if mask<0
#pragma unroll
    for (int r = 0; r < 9; ++r)
        Wk[(size_t)co * KDIM + r * CIN + ci] =
            (_Float16)(W[(size_t)co * KDIM + ci * 9 + r] * g);
}

// ---------------------------------------------------------------------------
// Main implicit-GEMM kernel. USE_WS: A tile streamed by the Tensor Data Mover
// (one tensor_load_to_lds per stage, LDS padding done by the D#); else f32
// stride-9 gather fallback.
// ---------------------------------------------------------------------------
template<bool USE_WS>
__global__ __launch_bounds__(256)
void binconv_wmma_f16(const float* __restrict__ X,
                      const float* __restrict__ Wt,
                      const float* __restrict__ Mask,
                      const _Float16* __restrict__ Wk,
                      float* __restrict__ Out)
{
    __shared__ _Float16 sA[2][BM][ROWH];   // gated weights  [co_row][k]
    __shared__ _Float16 sB[2][BN][ROWH];   // binarized im2col [pixel][k]

    const int tid  = threadIdx.x;
    const int lane = tid & 31;
    const int wave = tid >> 5;
    const int wm   = wave & 1;     // M offset = wm*64
    const int wn   = wave >> 1;    // N offset = wn*32

    const int coBase = blockIdx.y * BM;
    const int pBase  = blockIdx.x * BN;

    // ---- A fallback loader mapping: 128 rows x 32 halves, 2 threads per row
    const int rowA = tid >> 1;            // 0..127
    const int cA   = tid & 1;
    float gate = 1.f;
    if constexpr (!USE_WS) {
        const float m   = Mask[coBase + rowA];
        const float sgn = (m > 0.f) ? 1.f : ((m < 0.f) ? -1.f : 0.f);
        gate = 1.f + (sgn - 1.f) * 0.5f;
    }

    // ---- B loader mapping: one pixel per thread, 16 k-gathers
    const int pixB = tid & 127;
    const int kB   = (tid >> 7) * 16;     // 0 or 16
    const int p    = pBase + pixB;
    const int bImg = p / HW;
    const int pin  = p - bImg * HW;
    const int py   = pin / WID;
    const int px   = pin - py * WID;
    const float* gX = X + (size_t)bImg * CIN * HW + pin;

    uint16_t rb[16];
    float    raf[16];     // only used when !USE_WS

    // ---- Tensor Data Mover: one 128x(64B) tile per stage, hw row padding
    auto tdmA = [&](int kt, int buf) {
        if constexpr (USE_WS) {
            uint64_t ga  = (uint64_t)(uintptr_t)Wk +
                           ((uint64_t)coBase * KDIM + (uint64_t)kt * BK) * 2;
            uint32_t lds = (uint32_t)(uintptr_t)&sA[buf][0][0];
            // D# group 0: count=1 | lds_addr | global_addr | type=2
            u32x4 g0 = { 1u, lds, (uint32_t)ga,
                         (uint32_t)(ga >> 32) | (2u << 30) };
            // D# group 1: data_size=2B, pad_enable, pad after 16 DW by 4 DW,
            // tensor_dim0=2304, tensor_dim1=256, tile=32x128, stride0=2304
            u32x8 g1 = { (1u << 16) | (1u << 20) | (3u << 22) | (3u << 25),
                         (uint32_t)KDIM << 16,
                         (uint32_t)COUT << 16,
                         (uint32_t)BK   << 16,
                         (uint32_t)BM,
                         (uint32_t)KDIM,
                         0u, 0u };
            u32x4 gz = { 0u, 0u, 0u, 0u };   // groups 2/3 unused (2D tile)
            asm volatile("tensor_load_to_lds %0, %1, %2, %3"
                         :: "s"(g0), "s"(g1), "s"(gz), "s"(gz)
                         : "memory");
        } else {
            const int r   = kt >> 3;             // tap 0..8
            const int ci0 = (kt & 7) * BK;
            const float* src = Wt + (size_t)(coBase + rowA) * KDIM +
                               (size_t)(ci0 + cA * 16) * 9 + r;
#pragma unroll
            for (int j = 0; j < 16; ++j) raf[j] = src[j * 9];
        }
    };

    auto gfetchB = [&](int kt) {
        const int r  = kt >> 3;                  // uniform tap
        const int kh = r / 3, kw = r - kh * 3;
        const int ci0 = (kt & 7) * BK + kB;
        const int yy = py + kh - 1, xx = px + kw - 1;
        const bool ok = ((unsigned)yy < WID) & ((unsigned)xx < WID);
        const float* src = gX + (size_t)ci0 * HW + (kh - 1) * WID + (kw - 1);
#pragma unroll
        for (int j = 0; j < 16; ++j) {
            float t = ok ? src[(size_t)j * HW] : 0.f;
            unsigned u = __float_as_uint(t);
            // sign(t) as raw f16 bits: +-1.0h, or 0 when t == +-0
            rb[j] = (u & 0x7fffffffu)
                        ? (uint16_t)(0x3C00u | ((u >> 16) & 0x8000u))
                        : (uint16_t)0;
        }
    };

    auto lstore = [&](int buf) {
        union { uint16_t u[16]; f32x4 q[2]; } ub;
#pragma unroll
        for (int i = 0; i < 16; ++i) ub.u[i] = rb[i];
        f32x4* db = (f32x4*)&sB[buf][pixB][kB];
        db[0] = ub.q[0]; db[1] = ub.q[1];
        if constexpr (!USE_WS) {
            union { _Float16 h[16]; f32x4 q[2]; } ua;
#pragma unroll
            for (int i = 0; i < 16; ++i) ua.h[i] = (_Float16)(raf[i] * gate);
            f32x4* da = (f32x4*)&sA[buf][rowA][cA * 16];
            da[0] = ua.q[0]; da[1] = ua.q[1];
        }
    };

    v8f acc[4][2];
#pragma unroll
    for (int mi = 0; mi < 4; ++mi)
#pragma unroll
        for (int ni = 0; ni < 2; ++ni) acc[mi][ni] = v8f{};

    const int frow = lane & 15;
    const int hsel = lane >> 4;

    // ---- prologue
    if constexpr (USE_WS) { if (wave == 0) tdmA(0, 0); }
    else                  { tdmA(0, 0); }
    gfetchB(0);
    if constexpr (USE_WS) { if (wave == 0) __builtin_amdgcn_s_wait_tensorcnt(0); }
    lstore(0);
    __syncthreads();

    for (int kt = 0; kt < NKT; ++kt) {
        const int  cur  = kt & 1;
        const bool more = (kt + 1 < NKT);
        if (more) {                       // buf cur^1 last read 1 barrier ago: safe
            if constexpr (USE_WS) { if (wave == 0) tdmA(kt + 1, cur ^ 1); }
            else                  { tdmA(kt + 1, cur ^ 1); }
            gfetchB(kt + 1);
        }

        v16h afr[4], bfr[2];
#pragma unroll
        for (int mi = 0; mi < 4; ++mi) {
            const f32x4* rp = (const f32x4*)&sA[cur][wm * 64 + mi * 16 + frow][0];
            FragU u; u.q[0] = rp[hsel]; u.q[1] = rp[2 + hsel];
            afr[mi] = u.v;
        }
#pragma unroll
        for (int ni = 0; ni < 2; ++ni) {
            const f32x4* rp = (const f32x4*)&sB[cur][wn * 32 + ni * 16 + frow][0];
            FragU u; u.q[0] = rp[hsel]; u.q[1] = rp[2 + hsel];
            bfr[ni] = u.v;
        }
#pragma unroll
        for (int mi = 0; mi < 4; ++mi)
#pragma unroll
            for (int ni = 0; ni < 2; ++ni)
                acc[mi][ni] = __builtin_amdgcn_wmma_f32_16x16x32_f16(
                    false, afr[mi], false, bfr[ni],
                    (short)0, acc[mi][ni], false, false);

        if (more) {
            if constexpr (USE_WS) { if (wave == 0) __builtin_amdgcn_s_wait_tensorcnt(0); }
            lstore(cur ^ 1);
            __syncthreads();              // single barrier per stage
        }
    }

    // ---- epilogue: VGPR r -> M = r + 8*hsel, N = lane%16
#pragma unroll
    for (int ni = 0; ni < 2; ++ni) {
        const int pcol = pBase + wn * 32 + ni * 16 + frow;
        const int bI = pcol / HW;
        const int pi = pcol - bI * HW;
        float* ob = Out + (size_t)bI * COUT * HW + pi;
#pragma unroll
        for (int mi = 0; mi < 4; ++mi) {
            const int co0 = coBase + wm * 64 + mi * 16 + hsel * 8;
#pragma unroll
            for (int r = 0; r < 8; ++r)
                ob[(size_t)(co0 + r) * HW] = acc[mi][ni][r];
        }
    }
}

extern "C" void kernel_launch(void* const* d_in, const int* in_sizes, int n_in,
                              void* d_out, int out_size, void* d_ws, size_t ws_size,
                              hipStream_t stream) {
    const float* x    = (const float*)d_in[0];   // [32,256,56,56]
    const float* w    = (const float*)d_in[1];   // [256,256,3,3]
    const float* mask = (const float*)d_in[2];   // [256]
    float* out = (float*)d_out;                  // [32,256,56,56]

    const size_t wkBytes = (size_t)COUT * KDIM * sizeof(_Float16);  // 1.15 MB
    dim3 grid(NPIX / BN, COUT / BM);             // (784, 2)
    dim3 block(256);                             // 8 wave32s

    if (d_ws != nullptr && ws_size >= wkBytes) {
        _Float16* wk = (_Float16*)d_ws;
        prep_weights<<<dim3(COUT), dim3(CIN), 0, stream>>>(w, mask, wk);
        binconv_wmma_f16<true><<<grid, block, 0, stream>>>(x, w, mask, wk, out);
    } else {
        binconv_wmma_f16<false><<<grid, block, 0, stream>>>(x, w, mask, nullptr, out);
    }
}
